// MultiModalAttention_76673756168631
// MI455X (gfx1250) — matrix operand order repaired and checked by reading, not compiled, not run on hardware
//
#include <hip/hip_runtime.h>
#include <hip/hip_bf16.h>

// ---------------------------------------------------------------------------
// MultiModalAttention on MI455X (gfx1250, wave32)
//   Pre-pass: fp32 -> bf16 conversion of activations, convert+transpose of
//             weights to [N,K] bf16 (one-time, bandwidth-cheap).
//   4 GEMMs:  v_wmma_f32_16x16x32_bf16, LDS double-buffered; tiles staged
//             with GLOBAL_LOAD_ASYNC_TO_LDS_B128 (builtin confirmed present;
//             signature takes v4i pointers in AS1/AS3), else uint4 copies.
//   Tiny VALU kernels for the 2-token attention and LayerNorm+residual.
// ---------------------------------------------------------------------------

typedef unsigned short u16;
typedef __attribute__((ext_vector_type(16))) __bf16 v16bf;
typedef __attribute__((ext_vector_type(16))) u16    v16u;
typedef __attribute__((ext_vector_type(8)))  u16    v8u;
typedef __attribute__((ext_vector_type(8)))  float  v8f;
typedef int v4i __attribute__((vector_size(16)));

#define BM 128
#define BN 128
#define BK 32
#define LDS_STRIDE 40   // 32 + 8 pad (u16) -> 80 bytes/row, multiple of 16B

#if __has_builtin(__builtin_amdgcn_global_load_async_to_lds_b128) && \
    __has_builtin(__builtin_amdgcn_s_wait_asynccnt)
#define USE_ASYNC_LDS 1
#else
#define USE_ASYNC_LDS 0
#endif

__device__ __forceinline__ u16 f2bf(float f) {
    unsigned u = __builtin_bit_cast(unsigned, f);
    u += 0x7FFFu + ((u >> 16) & 1u);   // round-to-nearest-even
    return (u16)(u >> 16);
}
__device__ __forceinline__ float bf2f(u16 h) {
    unsigned u = ((unsigned)h) << 16;
    return __builtin_bit_cast(float, u);
}

union Frag { v16u v; v8u h[2]; };

// Copy one 16B chunk of a bf16 tile row from global into (padded) LDS.
__device__ __forceinline__ void stage_chunk(const u16* __restrict__ g, u16* l) {
#if USE_ASYNC_LDS
    __builtin_amdgcn_global_load_async_to_lds_b128(
        (__attribute__((address_space(1))) v4i*)(g),
        (__attribute__((address_space(3))) v4i*)(l),
        0, 0);
#else
    *(uint4*)l = *(const uint4*)g;
#endif
}

// C[M,N](+bias) = A[M,K] @ Bt[N,K]^T ; A,Bt bf16 row-major, C f32 or bf16.
// M,N multiples of 128; K multiple of 32.
__global__ __launch_bounds__(256) void gemm_bf16_wmma(
    const u16* __restrict__ A, const u16* __restrict__ Bt,
    const float* __restrict__ bias, void* __restrict__ Cout,
    int M, int N, int K, int ldc, int out_is_bf16)
{
    __shared__ __align__(16) u16 As[2][BM * LDS_STRIDE];   // [m][k]
    __shared__ __align__(16) u16 Bs[2][BN * LDS_STRIDE];   // [n][k]

    const int tid  = threadIdx.x;
    const int wave = tid >> 5;
    const int lane = tid & 31;
    const int half = lane >> 4;     // K sub-block selector (16-bit A/B layouts)
    const int r    = lane & 15;     // row (A) / col (B) within 16x16 tile

    const int bm = blockIdx.y * BM;
    const int bn = blockIdx.x * BN;
    const int wm = (wave & 1) * 64;       // 2 waves along M
    const int wn = (wave >> 1) * 32;      // 4 waves along N

    // Each thread stages 2 chunks per tile: chunk id -> (row, 16B chunk c)
    const int st_row  = (tid + 0)   >> 2;         // ids tid and tid+256
    const int st_c0   = (tid & 3) * 8;            // u16 offset within row
    const int st_row1 = (tid + 256) >> 2;

    v8f acc[4][2];
#pragma unroll
    for (int i = 0; i < 4; ++i)
#pragma unroll
        for (int j = 0; j < 2; ++j) acc[i][j] = v8f{};

    // ---- prologue: stage tile 0 into buffer 0 ----
    {
        const u16* gA = A + (size_t)bm * K;
        const u16* gB = Bt + (size_t)bn * K;
        stage_chunk(gA + (size_t)st_row  * K + st_c0, &As[0][st_row  * LDS_STRIDE + st_c0]);
        stage_chunk(gA + (size_t)st_row1 * K + st_c0, &As[0][st_row1 * LDS_STRIDE + st_c0]);
        stage_chunk(gB + (size_t)st_row  * K + st_c0, &Bs[0][st_row  * LDS_STRIDE + st_c0]);
        stage_chunk(gB + (size_t)st_row1 * K + st_c0, &Bs[0][st_row1 * LDS_STRIDE + st_c0]);
    }
#if USE_ASYNC_LDS
    __builtin_amdgcn_s_wait_asynccnt(0);
#endif
    __syncthreads();

    int cur = 0;
    for (int k0 = 0; k0 < K; k0 += BK) {
        // ---- issue async stage of next K-tile into the other buffer ----
        if (k0 + BK < K) {
            const u16* gA = A + (size_t)bm * K + (k0 + BK);
            const u16* gB = Bt + (size_t)bn * K + (k0 + BK);
            u16* lA = &As[cur ^ 1][0];
            u16* lB = &Bs[cur ^ 1][0];
            stage_chunk(gA + (size_t)st_row  * K + st_c0, lA + st_row  * LDS_STRIDE + st_c0);
            stage_chunk(gA + (size_t)st_row1 * K + st_c0, lA + st_row1 * LDS_STRIDE + st_c0);
            stage_chunk(gB + (size_t)st_row  * K + st_c0, lB + st_row  * LDS_STRIDE + st_c0);
            stage_chunk(gB + (size_t)st_row1 * K + st_c0, lB + st_row1 * LDS_STRIDE + st_c0);
        }
        // speculative prefetch two tiles ahead (global_prefetch_b8)
        if (k0 + 2 * BK < K) {
            __builtin_prefetch(A + (size_t)(bm + st_row) * K + k0 + 2 * BK, 0, 1);
            __builtin_prefetch(Bt + (size_t)(bn + st_row) * K + k0 + 2 * BK, 0, 1);
        }

        // ---- fragments from current buffer ----
        // A 16x32: lanes 0-15 row M=r hold K {0..7,16..23}; lanes 16-31 {8..15,24..31}
        const u16* lA = &As[cur][0];
        const u16* lB = &Bs[cur][0];
        Frag af[4];
#pragma unroll
        for (int mi = 0; mi < 4; ++mi) {
            const u16* base = lA + (wm + mi * 16 + r) * LDS_STRIDE;
            af[mi].h[0] = *(const v8u*)(base + half * 8);
            af[mi].h[1] = *(const v8u*)(base + 16 + half * 8);
        }
        // B 32x16: lanes 0-15 col N=r hold K {0..7,16..23}; lanes 16-31 {8..15,24..31}
        Frag bfr[2];
#pragma unroll
        for (int nj = 0; nj < 2; ++nj) {
            const u16* base = lB + (wn + nj * 16 + r) * LDS_STRIDE;
            bfr[nj].h[0] = *(const v8u*)(base + half * 8);
            bfr[nj].h[1] = *(const v8u*)(base + 16 + half * 8);
        }

#pragma unroll
        for (int mi = 0; mi < 4; ++mi)
#pragma unroll
            for (int nj = 0; nj < 2; ++nj)
                acc[mi][nj] = __builtin_amdgcn_wmma_f32_16x16x32_bf16(
                    false, __builtin_bit_cast(v16bf, af[mi].v),
                    false, __builtin_bit_cast(v16bf, bfr[nj].v),
                    (short)0, acc[mi][nj], false, false);

#if USE_ASYNC_LDS
        __builtin_amdgcn_s_wait_asynccnt(0);
#endif
        __syncthreads();
        cur ^= 1;
    }

    // ---- epilogue: C/D layout: VGPR v, lanes 0-15 M=v, lanes 16-31 M=8+v ----
#pragma unroll
    for (int mi = 0; mi < 4; ++mi)
#pragma unroll
        for (int nj = 0; nj < 2; ++nj) {
            const int col = bn + wn + nj * 16 + r;
            const float bb = bias ? bias[col] : 0.0f;
            if (out_is_bf16) {
                u16* C = (u16*)Cout;
#pragma unroll
                for (int v = 0; v < 8; ++v) {
                    const int row = bm + wm + mi * 16 + half * 8 + v;
                    C[(size_t)row * ldc + col] = f2bf(acc[mi][nj][v] + bb);
                }
            } else {
                float* C = (float*)Cout;
#pragma unroll
                for (int v = 0; v < 8; ++v) {
                    const int row = bm + wm + mi * 16 + half * 8 + v;
                    C[(size_t)row * ldc + col] = acc[mi][nj][v] + bb;
                }
            }
        }
}

// ---------------------------------------------------------------------------
// fp32 -> bf16 elementwise (n4 = element count / 4)
// ---------------------------------------------------------------------------
__global__ __launch_bounds__(256) void conv_bf16_kernel(
    const float* __restrict__ x, u16* __restrict__ y, long long n4)
{
    const long long i = (long long)blockIdx.x * 256 + threadIdx.x;
    if (i >= n4) return;
    const float4 f = *(const float4*)(x + i * 4);
    uint2 p;
    p.x = (unsigned)f2bf(f.x) | ((unsigned)f2bf(f.y) << 16);
    p.y = (unsigned)f2bf(f.z) | ((unsigned)f2bf(f.w) << 16);
    *(uint2*)(y + i * 4) = p;
}

// ---------------------------------------------------------------------------
// W[K,N] fp32 -> Wt[N,K] bf16 (32x32 LDS tiles, 256 threads)
// ---------------------------------------------------------------------------
__global__ __launch_bounds__(256) void transpose_conv_kernel(
    const float* __restrict__ W, u16* __restrict__ Wt, int K, int N)
{
    __shared__ float t[32][33];
    const int bn = blockIdx.x * 32;        // N origin
    const int bk = blockIdx.y * 32;        // K origin
    const int tx = threadIdx.x & 31;
    const int ty = threadIdx.x >> 5;       // 8 rows per pass
#pragma unroll
    for (int i = 0; i < 32; i += 8)
        t[ty + i][tx] = W[(size_t)(bk + ty + i) * N + bn + tx];
    __syncthreads();
#pragma unroll
    for (int i = 0; i < 32; i += 8)
        Wt[(size_t)(bn + ty + i) * K + bk + tx] = f2bf(t[tx][ty + i]);
}

// ---------------------------------------------------------------------------
// 2-token attention, bf16 in/out: one thread per (b, head).
// qkv is [2B, 3H] bf16 row-major; out is [2B, H] bf16.
// ---------------------------------------------------------------------------
__device__ __forceinline__ void ld4bf(const u16* p, float* f) {
    const uint2 w = *(const uint2*)p;
    f[0] = bf2f((u16)(w.x & 0xffffu)); f[1] = bf2f((u16)(w.x >> 16));
    f[2] = bf2f((u16)(w.y & 0xffffu)); f[3] = bf2f((u16)(w.y >> 16));
}

__global__ __launch_bounds__(256) void attn2_kernel(
    const u16* __restrict__ qkv, u16* __restrict__ out, int B)
{
    const int t = blockIdx.x * blockDim.x + threadIdx.x;
    if (t >= B * 16) return;
    const int b = t >> 4, h = t & 15;
    const float scale = 0.125f;   // 64^-0.5

    const u16* q0 = qkv + (size_t)(2 * b) * 3072 + h * 64;
    const u16* q1 = q0 + 3072;
    const u16* k0 = q0 + 1024;
    const u16* k1 = k0 + 3072;
    const u16* v0 = q0 + 2048;
    const u16* v1 = v0 + 3072;

    float s00 = 0.f, s01 = 0.f, s10 = 0.f, s11 = 0.f;
#pragma unroll 4
    for (int d = 0; d < 64; d += 4) {
        float a0[4], a1[4], c0[4], c1[4];
        ld4bf(q0 + d, a0); ld4bf(q1 + d, a1);
        ld4bf(k0 + d, c0); ld4bf(k1 + d, c1);
#pragma unroll
        for (int j = 0; j < 4; ++j) {
            s00 += a0[j] * c0[j]; s01 += a0[j] * c1[j];
            s10 += a1[j] * c0[j]; s11 += a1[j] * c1[j];
        }
    }
    s00 *= scale; s01 *= scale; s10 *= scale; s11 *= scale;

    const float m0 = fmaxf(s00, s01), m1 = fmaxf(s10, s11);
    const float e00 = __expf(s00 - m0), e01 = __expf(s01 - m0);
    const float e10 = __expf(s10 - m1), e11 = __expf(s11 - m1);
    const float z0 = 1.f / (e00 + e01), z1 = 1.f / (e10 + e11);
    const float a00 = e00 * z0, a01 = e01 * z0;
    const float a10 = e10 * z1, a11 = e11 * z1;

    u16* o0 = out + (size_t)(2 * b) * 1024 + h * 64;
    u16* o1 = o0 + 1024;
#pragma unroll 4
    for (int d = 0; d < 64; d += 4) {
        float x0[4], x1[4];
        ld4bf(v0 + d, x0); ld4bf(v1 + d, x1);
        uint2 p0, p1;
        p0.x = (unsigned)f2bf(a00 * x0[0] + a01 * x1[0]) |
               ((unsigned)f2bf(a00 * x0[1] + a01 * x1[1]) << 16);
        p0.y = (unsigned)f2bf(a00 * x0[2] + a01 * x1[2]) |
               ((unsigned)f2bf(a00 * x0[3] + a01 * x1[3]) << 16);
        p1.x = (unsigned)f2bf(a10 * x0[0] + a11 * x1[0]) |
               ((unsigned)f2bf(a10 * x0[1] + a11 * x1[1]) << 16);
        p1.y = (unsigned)f2bf(a10 * x0[2] + a11 * x1[2]) |
               ((unsigned)f2bf(a10 * x0[3] + a11 * x1[3]) << 16);
        *(uint2*)(o0 + d) = p0;
        *(uint2*)(o1 + d) = p1;
    }
}

// ---------------------------------------------------------------------------
// In-place LayerNorm(2048) + residual: y <- LN(y)*gamma+beta + bf2f(res)
// ---------------------------------------------------------------------------
__global__ __launch_bounds__(256) void ln_res_kernel(
    float* __restrict__ y, const u16* __restrict__ res,
    const float* __restrict__ gamma, const float* __restrict__ beta)
{
    __shared__ float ssum[256];
    __shared__ float ssq[256];
    const int b = blockIdx.x;
    const int tid = threadIdx.x;
    const size_t base = (size_t)b * 2048;

    float vals[8];
    float s = 0.f, sq = 0.f;
#pragma unroll
    for (int i = 0; i < 8; ++i) {
        const float x = y[base + tid + i * 256];
        vals[i] = x; s += x; sq += x * x;
    }
    ssum[tid] = s; ssq[tid] = sq;
    __syncthreads();
    for (int off = 128; off > 0; off >>= 1) {
        if (tid < off) { ssum[tid] += ssum[tid + off]; ssq[tid] += ssq[tid + off]; }
        __syncthreads();
    }
    const float mu  = ssum[0] * (1.f / 2048.f);
    const float var = ssq[0] * (1.f / 2048.f) - mu * mu;
    const float inv = rsqrtf(var + 1e-5f);
#pragma unroll
    for (int i = 0; i < 8; ++i) {
        const int c = tid + i * 256;
        y[base + c] = (vals[i] - mu) * inv * gamma[c] + beta[c] + bf2f(res[base + c]);
    }
}

// ---------------------------------------------------------------------------
// Inputs (setup_inputs order):
//  0 features1 [8192,2048]  1 features2 [8192,768]
//  2 W1 [2048,1024]  3 b1 [1024]  4 W2 [768,1024]  5 b2 [1024]
//  6 Wqkv [1024,3072]  7 Wout [1024,1024]  8 bout [1024]
//  9 gamma [2048]  10 beta [2048]
// Output: [8192, 2048] fp32
// ---------------------------------------------------------------------------
extern "C" void kernel_launch(void* const* d_in, const int* in_sizes, int n_in,
                              void* d_out, int out_size, void* d_ws, size_t ws_size,
                              hipStream_t stream) {
    const float* f1   = (const float*)d_in[0];
    const float* f2   = (const float*)d_in[1];
    const float* W1   = (const float*)d_in[2];
    const float* b1   = (const float*)d_in[3];
    const float* W2   = (const float*)d_in[4];
    const float* b2   = (const float*)d_in[5];
    const float* Wqkv = (const float*)d_in[6];
    const float* Wout = (const float*)d_in[7];
    const float* bout = (const float*)d_in[8];
    const float* gamma= (const float*)d_in[9];
    const float* beta = (const float*)d_in[10];

    const int Bn = 8192, H = 1024;

    // workspace layout (bf16 / u16 elements)
    u16* A1    = (u16*)d_ws;                          // [8192, 2048]
    u16* A2    = A1    + (size_t)Bn * 2048;           // [8192, 768]
    u16* W1t   = A2    + (size_t)Bn * 768;            // [1024, 2048]
    u16* W2t   = W1t   + (size_t)1024 * 2048;         // [1024, 768]
    u16* Wqkvt = W2t   + (size_t)1024 * 768;          // [3072, 1024]
    u16* Woutt = Wqkvt + (size_t)3072 * 1024;         // [1024, 1024]
    u16* xres  = Woutt + (size_t)1024 * 1024;         // [8192, 2048] = x & residual
    u16* qkvb  = xres  + (size_t)Bn * 2048;           // [16384, 3072]
    u16* attn  = qkvb  + (size_t)2 * Bn * 3072;       // [16384, 1024]
    float* y   = (float*)d_out;                       // [16384, 1024] view of out

    dim3 blk(256);

    // --- pre-pass: activations to bf16, weights to transposed bf16 ---
    {
        const long long n4a = (long long)Bn * 2048 / 4;
        conv_bf16_kernel<<<(unsigned)((n4a + 255) / 256), blk, 0, stream>>>(f1, A1, n4a);
        const long long n4b = (long long)Bn * 768 / 4;
        conv_bf16_kernel<<<(unsigned)((n4b + 255) / 256), blk, 0, stream>>>(f2, A2, n4b);
    }
    transpose_conv_kernel<<<dim3(1024 / 32, 2048 / 32), blk, 0, stream>>>(W1, W1t, 2048, 1024);
    transpose_conv_kernel<<<dim3(1024 / 32,  768 / 32), blk, 0, stream>>>(W2, W2t,  768, 1024);
    transpose_conv_kernel<<<dim3(3072 / 32, 1024 / 32), blk, 0, stream>>>(Wqkv, Wqkvt, 1024, 3072);
    transpose_conv_kernel<<<dim3(1024 / 32, 1024 / 32), blk, 0, stream>>>(Wout, Woutt, 1024, 1024);

    // --- f1/f2 projections -> xres[:,0:1024] / xres[:,1024:2048] (bf16) ---
    gemm_bf16_wmma<<<dim3(H / BN, Bn / BM), blk, 0, stream>>>(
        A1, W1t, b1, xres,      Bn, H, 2048, 2 * H, 1);
    gemm_bf16_wmma<<<dim3(H / BN, Bn / BM), blk, 0, stream>>>(
        A2, W2t, b2, xres + H,  Bn, H,  768, 2 * H, 1);
    // --- qkv = x @ Wqkv (x == xres flat-viewed as [16384, 1024]) ---
    gemm_bf16_wmma<<<dim3(3 * H / BN, 2 * Bn / BM), blk, 0, stream>>>(
        xres, Wqkvt, nullptr, qkvb, 2 * Bn, 3 * H, H, 3 * H, 1);
    // --- 2-token softmax attention ---
    attn2_kernel<<<(Bn * 16 + 255) / 256, blk, 0, stream>>>(qkvb, attn, Bn);
    // --- out = attn @ Wout + bout -> d_out (fp32) ---
    gemm_bf16_wmma<<<dim3(H / BN, 2 * Bn / BM), blk, 0, stream>>>(
        attn, Woutt, bout, y, 2 * Bn, H, H, H, 0);
    // --- LayerNorm(2048) + residual, in place on d_out ---
    ln_res_kernel<<<Bn, blk, 0, stream>>>(y, xres, gamma, beta);
}
